// IJEPAModel_90117003804944
// MI455X (gfx1250) — matrix operand order repaired and checked by reading, not compiled, not run on hardware
//
#include <hip/hip_runtime.h>

// ---------------------------------------------------------------------------
// I-JEPA (ViT-Base ctx/tgt + predictor + decoder) for gfx1250 (MI455X).
// GEMMs: V_WMMA_F32_16X16X32_BF16, fp32 accumulate.  Block = 256 thr (8
// wave32), C tile 128x128, wave tile 32x64 (8 WMMA / 32-K step), K chunk 64.
// Weight tiles staged global->LDS with GLOBAL_LOAD_ASYNC_TO_LDS_B128 when the
// toolchain exposes it (ASYNCcnt + s_wait_asynccnt), else sync b128 copies.
// Activations converted fp32->bf16 while staging to LDS.
// ---------------------------------------------------------------------------

typedef __attribute__((ext_vector_type(16))) __bf16 v16bf;
typedef __attribute__((ext_vector_type(8)))  float  v8f;
typedef int v4i32 __attribute__((vector_size(16)));

#define AS1 __attribute__((address_space(1)))
#define AS3 __attribute__((address_space(3)))

#ifndef __has_builtin
#define __has_builtin(x) 0
#endif
#if __has_builtin(__builtin_amdgcn_global_load_async_to_lds_b128)
#define USE_ASYNC_LDS 1
#else
#define USE_ASYNC_LDS 0
#endif

#define BATCH   16
#define DM      768
#define NLAYER  12
#define NHEAD   12
#define PP      8
#define IMGS    96
#define GRIDP   12          // IMGS / PP
#define NPATCH  144         // GRIDP^2
#define SEQ     145         // NPATCH + 1
#define HDIM    64          // DM / NHEAD
#define MLPD    3072
#define MTOK    (BATCH*SEQ) // 2320
#define PATCHE  192         // 3*PP*PP
#define LNEPS   1e-6f

__device__ __forceinline__ unsigned short f2bf(float f) {
    unsigned u = __builtin_bit_cast(unsigned, f);
    u += 0x7FFFu + ((u >> 16) & 1u);          // round-to-nearest-even
    return (unsigned short)(u >> 16);
}

// ---------------------------- fp32 -> bf16 convert --------------------------
__global__ void cvt_f32_bf16_k(const float* __restrict__ src,
                               unsigned short* __restrict__ dst, int n) {
    int i = (blockIdx.x * blockDim.x + threadIdx.x) * 4;
    if (i + 3 < n) {
        float4 f = *reinterpret_cast<const float4*>(src + i);
        dst[i + 0] = f2bf(f.x); dst[i + 1] = f2bf(f.y);
        dst[i + 2] = f2bf(f.z); dst[i + 3] = f2bf(f.w);
    } else {
        for (; i < n; ++i) dst[i] = f2bf(src[i]);
    }
}

// ------------------------------- WMMA GEMM ----------------------------------
// C[M,N] = act(A[M,K] @ W[N,K]^T + bias) (+ res).  A fp32, W bf16(N,K) row
// major, C fp32.  K % 64 == 0.  Block: 256 thr = 8 waves arranged 4(M) x 2(N);
// each wave owns a 32x64 C tile = 2x4 16x16 fragments.
template <int ACT, bool RES>
__global__ __launch_bounds__(256)
void gemm_bf16_wmma_k(const float* __restrict__ A,
                      const unsigned short* __restrict__ W,
                      const float* __restrict__ bias,
                      const float* __restrict__ res,
                      float* __restrict__ C, int M, int N, int K) {
    __shared__ unsigned int ldsA[128][32];   // 128 rows x 64 bf16 (16 KB)
    __shared__ unsigned int ldsW[128][32];   // 128 cols x 64 bf16 (16 KB)

    const int t    = threadIdx.x;
    const int lane = t & 31, wid = t >> 5;
    const int wm = wid >> 1, wn = wid & 1;
    const int g = lane >> 4, l15 = lane & 15;
    const int row0 = blockIdx.y * 128, col0 = blockIdx.x * 128;

    v8f acc[2][4] = {};

    union Frag { v16bf v; unsigned int u[8]; };

    for (int k0 = 0; k0 < K; k0 += 64) {
        // ---- stage W tile (bf16, pure copy): 128 x 64 bf16 = 1024 x b128
#pragma unroll
        for (int i = 0; i < 4; ++i) {
            const int idx = t + i * 256;
            const int n = idx >> 3, ks = (idx & 7) * 8;   // bf16 elem offset
            if (col0 + n < N) {
                const unsigned short* gp = W + (size_t)(col0 + n) * K + k0 + ks;
#if USE_ASYNC_LDS
                __builtin_amdgcn_global_load_async_to_lds_b128(
                    (AS1 v4i32*)(unsigned long long)(size_t)gp,
                    (AS3 v4i32*)&ldsW[n][ks >> 1], 0, 0);
#else
                *reinterpret_cast<uint4*>(&ldsW[n][ks >> 1]) =
                    *reinterpret_cast<const uint4*>(gp);
#endif
            }
        }
        // ---- stage A tile (fp32 -> bf16 convert): 128 rows x 64 k
#pragma unroll
        for (int i = 0; i < 4; ++i) {
            const int idx = t + i * 256;
            const int r = idx >> 3, ks = (idx & 7) * 8;
            const int gr = row0 + r;
            float v[8];
            if (gr < M) {
                const float4* p =
                    reinterpret_cast<const float4*>(A + (size_t)gr * K + k0 + ks);
                float4 f0 = p[0], f1 = p[1];
                v[0] = f0.x; v[1] = f0.y; v[2] = f0.z; v[3] = f0.w;
                v[4] = f1.x; v[5] = f1.y; v[6] = f1.z; v[7] = f1.w;
                if (k0 + 64 < K)
                    __builtin_prefetch(A + (size_t)gr * K + k0 + 64 + ks, 0, 1);
            } else {
#pragma unroll
                for (int q = 0; q < 8; ++q) v[q] = 0.f;
            }
            uint4 u;
            u.x = (unsigned)f2bf(v[0]) | ((unsigned)f2bf(v[1]) << 16);
            u.y = (unsigned)f2bf(v[2]) | ((unsigned)f2bf(v[3]) << 16);
            u.z = (unsigned)f2bf(v[4]) | ((unsigned)f2bf(v[5]) << 16);
            u.w = (unsigned)f2bf(v[6]) | ((unsigned)f2bf(v[7]) << 16);
            *reinterpret_cast<uint4*>(&ldsA[r][ks >> 1]) = u;
        }
#if USE_ASYNC_LDS
#if __has_builtin(__builtin_amdgcn_s_wait_asynccnt)
        __builtin_amdgcn_s_wait_asynccnt(0);
#else
        asm volatile("s_wait_asynccnt 0" ::: "memory");
#endif
#endif
        __syncthreads();

        // ---- two 32-K sub-steps
#pragma unroll
        for (int kk = 0; kk < 2; ++kk) {
            const int kb = kk * 16;            // uint offset of sub-chunk
            Frag a[2];
#pragma unroll
            for (int am = 0; am < 2; ++am) {
                const int m = wm * 32 + am * 16 + l15;
#pragma unroll
                for (int i = 0; i < 8; ++i)    // ISA 16-bit A 16x32 layout
                    a[am].u[i] = ldsA[m][kb + 4 * g + (i < 4 ? i : 4 + i)];
            }
#pragma unroll
            for (int j = 0; j < 4; ++j) {
                Frag b;
                const int n = wn * 64 + j * 16 + l15;
#pragma unroll
                for (int i = 0; i < 8; ++i) b.u[i] = ldsW[n][kb + 4 * g + (i < 4 ? i : 4 + i)];
#pragma unroll
                for (int am = 0; am < 2; ++am)
                    acc[am][j] = __builtin_amdgcn_wmma_f32_16x16x32_bf16(
                        false, a[am].v, false, b.v, (short)0, acc[am][j],
                        false, false);
            }
        }
        __syncthreads();
    }

    // ---- epilogue: bias, activation, residual
#pragma unroll
    for (int j = 0; j < 4; ++j) {
        const int n = col0 + wn * 64 + j * 16 + l15;
        if (n < N) {
            const float bv = bias ? bias[n] : 0.f;
#pragma unroll
            for (int am = 0; am < 2; ++am) {
#pragma unroll
                for (int v = 0; v < 8; ++v) {
                    const int m = row0 + wm * 32 + am * 16 + v + 8 * g;
                    if (m < M) {
                        float val = acc[am][j][v] + bv;
                        if (ACT == 1)
                            val = 0.5f * val * (1.0f + erff(val * 0.70710678118f));
                        else if (ACT == 2)
                            val = fmaxf(val, 0.f);
                        if (RES) val += res[(size_t)m * N + n];
                        C[(size_t)m * N + n] = val;
                    }
                }
            }
        }
    }
}

// ------------------------------ LayerNorm -----------------------------------
__global__ __launch_bounds__(256)
void layernorm_k(const float* __restrict__ x, const float* __restrict__ w,
                 const float* __restrict__ b, float* __restrict__ y, int M) {
    const int t = threadIdx.x, lane = t & 31, wid = t >> 5;
    const int row = blockIdx.x * 8 + wid;
    if (row >= M) return;
    const float4* xp = reinterpret_cast<const float4*>(x + (size_t)row * DM);
    float4 v[6];
    float s1 = 0.f, s2 = 0.f;
#pragma unroll
    for (int i = 0; i < 6; ++i) {
        v[i] = xp[lane + i * 32];
        s1 += v[i].x + v[i].y + v[i].z + v[i].w;
        s2 += v[i].x * v[i].x + v[i].y * v[i].y + v[i].z * v[i].z + v[i].w * v[i].w;
    }
#pragma unroll
    for (int off = 16; off; off >>= 1) {
        s1 += __shfl_xor(s1, off, 32);
        s2 += __shfl_xor(s2, off, 32);
    }
    const float mean = s1 * (1.f / DM);
    const float var  = s2 * (1.f / DM) - mean * mean;
    const float rs   = rsqrtf(var + LNEPS);
    const float4* wp = reinterpret_cast<const float4*>(w);
    const float4* bp = reinterpret_cast<const float4*>(b);
    float4* yp = reinterpret_cast<float4*>(y + (size_t)row * DM);
#pragma unroll
    for (int i = 0; i < 6; ++i) {
        float4 wv = wp[lane + i * 32], bv = bp[lane + i * 32], o;
        o.x = (v[i].x - mean) * rs * wv.x + bv.x;
        o.y = (v[i].y - mean) * rs * wv.y + bv.y;
        o.z = (v[i].z - mean) * rs * wv.z + bv.z;
        o.w = (v[i].w - mean) * rs * wv.w + bv.w;
        yp[lane + i * 32] = o;
    }
}

// ------------------------------ Attention -----------------------------------
// One wave per (b, h, q).  S=145, HD=64.  qkv: [M=2320][2304] (q|k|v).
__global__ __launch_bounds__(128)
void attention_k(const float* __restrict__ qkv, float* __restrict__ o) {
    __shared__ float qs[4][HDIM];
    __shared__ float ps[4][160];
    const int t = threadIdx.x, lane = t & 31, wid = t >> 5;
    const int w = blockIdx.x * 4 + wid;       // exact: B*H*S % 4 == 0
    const int q = w % SEQ;
    const int bh = w / SEQ;
    const int h = bh % NHEAD, b = bh / NHEAD;
    const float scale = 0.125f;               // 1/sqrt(64)

    const size_t qrow = ((size_t)(b * SEQ + q)) * (3 * DM) + h * HDIM;
    qs[wid][lane * 2 + 0] = qkv[qrow + lane * 2 + 0];
    qs[wid][lane * 2 + 1] = qkv[qrow + lane * 2 + 1];
    __syncthreads();

    float mx = -1e30f;
    for (int kk = lane; kk < SEQ; kk += 32) {
        const float4* kp = reinterpret_cast<const float4*>(
            qkv + ((size_t)(b * SEQ + kk)) * (3 * DM) + DM + h * HDIM);
        float dot = 0.f;
#pragma unroll
        for (int i = 0; i < 16; ++i) {
            float4 kv = kp[i];
            dot += qs[wid][4 * i + 0] * kv.x + qs[wid][4 * i + 1] * kv.y +
                   qs[wid][4 * i + 2] * kv.z + qs[wid][4 * i + 3] * kv.w;
        }
        float s = dot * scale;
        ps[wid][kk] = s;
        mx = fmaxf(mx, s);
    }
#pragma unroll
    for (int off = 16; off; off >>= 1) mx = fmaxf(mx, __shfl_xor(mx, off, 32));
    __syncthreads();

    float lsum = 0.f;
    for (int kk = lane; kk < SEQ; kk += 32) {
        float e = expf(ps[wid][kk] - mx);
        ps[wid][kk] = e;
        lsum += e;
    }
#pragma unroll
    for (int off = 16; off; off >>= 1) lsum += __shfl_xor(lsum, off, 32);
    __syncthreads();

    const float inv = 1.f / lsum;
    float o0 = 0.f, o1 = 0.f;
    const int d0 = lane, d1 = lane + 32;
    for (int kk = 0; kk < SEQ; ++kk) {
        const float p = ps[wid][kk];
        const float* vp = qkv + ((size_t)(b * SEQ + kk)) * (3 * DM) + 2 * DM + h * HDIM;
        o0 += p * vp[d0];
        o1 += p * vp[d1];
    }
    const size_t orow = ((size_t)(b * SEQ + q)) * DM + h * HDIM;
    o[orow + d0] = o0 * inv;
    o[orow + d1] = o1 * inv;
}

// ------------------------- masks / patches / glue ---------------------------
__global__ void keep_k(const int* __restrict__ ci, int* __restrict__ keep) {
    int i = blockIdx.x * blockDim.x + threadIdx.x;
    if (i >= BATCH * NPATCH) return;
    int b = i / NPATCH, p = i % NPATCH, f = 0;
    for (int j = 0; j < NPATCH; ++j) f |= (ci[b * NPATCH + j] == p);
    keep[i] = f;
}

__global__ void replaced_k(const int* __restrict__ mk, int* __restrict__ rep) {
    int i = blockIdx.x * blockDim.x + threadIdx.x;
    if (i >= BATCH * NPATCH) return;
    int b = i / NPATCH, p = i % NPATCH, f = 0;
    for (int j = 0; j < 108; ++j) f |= (mk[b * 108 + j] == p);   // (4-1)*36
    rep[i] = f;
}

__global__ void im2col_k(const float* __restrict__ img, const int* __restrict__ keep,
                         float* __restrict__ out, int total) {
    int i = blockIdx.x * blockDim.x + threadIdx.x;
    if (i >= total) return;
    int e = i % PATCHE, p = (i / PATCHE) % NPATCH, b = i / (PATCHE * NPATCH);
    int c = e >> 6, r = (e >> 3) & 7, col = e & 7;
    int gy = p / GRIDP, gx = p % GRIDP;
    float v = img[(((size_t)b * 3 + c) * IMGS + gy * PP + r) * IMGS + gx * PP + col];
    if (keep && !keep[b * NPATCH + p]) v = 0.f;
    out[i] = v;
}

__global__ void assemble_k(const float* __restrict__ emb, const float* __restrict__ cls,
                           const float* __restrict__ pos, float* __restrict__ x) {
    int i = blockIdx.x * blockDim.x + threadIdx.x;
    if (i >= MTOK * DM) return;
    int d = i % DM, s = (i / DM) % SEQ, b = i / (DM * SEQ);
    float v = (s == 0) ? cls[d] : emb[((size_t)b * NPATCH + (s - 1)) * DM + d];
    x[i] = v + pos[s * DM + d];
}

__global__ void extract_cls_k(const float* __restrict__ y, float* __restrict__ out) {
    int i = blockIdx.x * blockDim.x + threadIdx.x;
    if (i >= BATCH * DM) return;
    int b = i / DM, d = i % DM;
    out[i] = y[((size_t)b * SEQ) * DM + d];
}

__global__ void addvec_k(const float* __restrict__ a, const float* __restrict__ mt,
                         float* __restrict__ o) {
    int i = blockIdx.x * blockDim.x + threadIdx.x;
    if (i >= BATCH * DM) return;
    o[i] = a[i] + mt[i % DM];
}

__global__ void reconstruct_k(const float* __restrict__ img, const float* __restrict__ pp,
                              const int* __restrict__ rep, float* __restrict__ out) {
    int i = blockIdx.x * blockDim.x + threadIdx.x;
    if (i >= BATCH * 3 * IMGS * IMGS) return;
    int x = i % IMGS, y = (i / IMGS) % IMGS, c = (i / (IMGS * IMGS)) % 3,
        b = i / (3 * IMGS * IMGS);
    int p = (y >> 3) * GRIDP + (x >> 3);
    float v = rep[b * NPATCH + p] ? pp[b * PATCHE + c * 64 + (y & 7) * 8 + (x & 7)]
                                  : img[i];
    out[i] = v;
}

// -------------------------------- driver ------------------------------------
extern "C" void kernel_launch(void* const* d_in, const int* in_sizes, int n_in,
                              void* d_out, int out_size, void* d_ws, size_t ws_size,
                              hipStream_t stream) {
    (void)in_sizes; (void)n_in; (void)out_size; (void)ws_size;

    const float* images  = (const float*)d_in[0];
    const int*   ctx_idx = (const int*)d_in[1];
    const int*   masks   = (const int*)d_in[2];
    // ctx params base 3, tgt params base 21 (conv_w, conv_b, cls, pos, lnf_w,
    // lnf_b, then layers: ln1_w, ln1_b, qkv_w, qkv_b, out_w, out_b, ln2_w,
    // ln2_b, fc1_w, fc1_b, fc2_w, fc2_b)
    const float* pred_w1 = (const float*)d_in[39];
    const float* pred_b1 = (const float*)d_in[40];
    const float* pred_w2 = (const float*)d_in[41];
    const float* pred_b2 = (const float*)d_in[42];
    const float* dec_w1  = (const float*)d_in[43];
    const float* dec_b1  = (const float*)d_in[44];
    const float* dec_w2  = (const float*)d_in[45];
    const float* dec_b2  = (const float*)d_in[46];
    const float* mask_tok= (const float*)d_in[47];

    // workspace carve
    char* wp = (char*)d_ws;
    auto carve = [&](size_t bytes) -> void* {
        void* p = (void*)wp;
        wp += (bytes + 255) & ~(size_t)255;
        return p;
    };
    int*   keep    = (int*)  carve((size_t)BATCH * NPATCH * 4);
    int*   rep     = (int*)  carve((size_t)BATCH * NPATCH * 4);
    float* col     = (float*)carve((size_t)BATCH * NPATCH * PATCHE * 4);
    float* embtmp  = (float*)carve((size_t)BATCH * NPATCH * DM * 4);
    float* xbuf    = (float*)carve((size_t)MTOK * DM * 4);
    float* ybuf    = (float*)carve((size_t)MTOK * DM * 4);
    float* obuf    = (float*)carve((size_t)MTOK * DM * 4);
    float* qkvbuf  = (float*)carve((size_t)MTOK * 3 * DM * 4);
    float* mlpbuf  = (float*)carve((size_t)MTOK * MLPD * 4);
    unsigned short* wbf = (unsigned short*)carve((size_t)MLPD * DM * 2);
    float* ptmp0   = (float*)carve((size_t)BATCH * DM * 4);
    float* ptmp1   = (float*)carve((size_t)BATCH * DM * 4);
    float* ppatch  = (float*)carve((size_t)BATCH * PATCHE * 4);

    float* out_ctx = (float*)d_out;
    float* out_tgt = out_ctx + (size_t)BATCH * DM;
    float* out_rec = out_tgt + (size_t)BATCH * DM;

    auto cvt = [&](const float* src, size_t n) {
        cvt_f32_bf16_k<<<(unsigned)((n / 4 + 255) / 256), 256, 0, stream>>>(
            src, wbf, (int)n);
    };
    auto gemm = [&](const float* A, const float* bias, const float* res,
                    float* C, int M, int N, int K, int act) {
        dim3 grid((N + 127) / 128, (M + 127) / 128), blk(256);
        if (res)           gemm_bf16_wmma_k<0, true ><<<grid, blk, 0, stream>>>(A, wbf, bias, res,     C, M, N, K);
        else if (act == 1) gemm_bf16_wmma_k<1, false><<<grid, blk, 0, stream>>>(A, wbf, bias, nullptr, C, M, N, K);
        else if (act == 2) gemm_bf16_wmma_k<2, false><<<grid, blk, 0, stream>>>(A, wbf, bias, nullptr, C, M, N, K);
        else               gemm_bf16_wmma_k<0, false><<<grid, blk, 0, stream>>>(A, wbf, bias, nullptr, C, M, N, K);
    };

    auto run_vit = [&](int base, const int* keepmask, float* emb_out) {
        const float* conv_w = (const float*)d_in[base + 0];
        const float* conv_b = (const float*)d_in[base + 1];
        const float* cls    = (const float*)d_in[base + 2];
        const float* pos    = (const float*)d_in[base + 3];
        const float* lnf_w  = (const float*)d_in[base + 4];
        const float* lnf_b  = (const float*)d_in[base + 5];
        const float* ln1_w  = (const float*)d_in[base + 6];
        const float* ln1_b  = (const float*)d_in[base + 7];
        const float* qkv_w  = (const float*)d_in[base + 8];
        const float* qkv_b  = (const float*)d_in[base + 9];
        const float* out_w  = (const float*)d_in[base + 10];
        const float* out_b  = (const float*)d_in[base + 11];
        const float* ln2_w  = (const float*)d_in[base + 12];
        const float* ln2_b  = (const float*)d_in[base + 13];
        const float* fc1_w  = (const float*)d_in[base + 14];
        const float* fc1_b  = (const float*)d_in[base + 15];
        const float* fc2_w  = (const float*)d_in[base + 16];
        const float* fc2_b  = (const float*)d_in[base + 17];

        const int tot_col = BATCH * NPATCH * PATCHE;
        im2col_k<<<(tot_col + 255) / 256, 256, 0, stream>>>(images, keepmask, col, tot_col);
        cvt(conv_w, (size_t)DM * PATCHE);
        gemm(col, conv_b, nullptr, embtmp, BATCH * NPATCH, DM, PATCHE, 0);
        const int tot_x = MTOK * DM;
        assemble_k<<<(tot_x + 255) / 256, 256, 0, stream>>>(embtmp, cls, pos, xbuf);

        const int ln_grid = (MTOK + 7) / 8;
        for (int l = 0; l < NLAYER; ++l) {
            layernorm_k<<<ln_grid, 256, 0, stream>>>(xbuf, ln1_w + l * DM, ln1_b + l * DM, ybuf, MTOK);
            cvt(qkv_w + (size_t)l * 3 * DM * DM, (size_t)3 * DM * DM);
            gemm(ybuf, qkv_b + l * 3 * DM, nullptr, qkvbuf, MTOK, 3 * DM, DM, 0);
            attention_k<<<(BATCH * NHEAD * SEQ) / 4, 128, 0, stream>>>(qkvbuf, obuf);
            cvt(out_w + (size_t)l * DM * DM, (size_t)DM * DM);
            gemm(obuf, out_b + l * DM, xbuf, xbuf, MTOK, DM, DM, 0);
            layernorm_k<<<ln_grid, 256, 0, stream>>>(xbuf, ln2_w + l * DM, ln2_b + l * DM, ybuf, MTOK);
            cvt(fc1_w + (size_t)l * MLPD * DM, (size_t)MLPD * DM);
            gemm(ybuf, fc1_b + l * MLPD, nullptr, mlpbuf, MTOK, MLPD, DM, 1);
            cvt(fc2_w + (size_t)l * DM * MLPD, (size_t)DM * MLPD);
            gemm(mlpbuf, fc2_b + l * DM, xbuf, xbuf, MTOK, DM, MLPD, 0);
        }
        layernorm_k<<<ln_grid, 256, 0, stream>>>(xbuf, lnf_w, lnf_b, ybuf, MTOK);
        extract_cls_k<<<(BATCH * DM + 255) / 256, 256, 0, stream>>>(ybuf, emb_out);
    };

    keep_k    <<<(BATCH * NPATCH + 255) / 256, 256, 0, stream>>>(ctx_idx, keep);
    replaced_k<<<(BATCH * NPATCH + 255) / 256, 256, 0, stream>>>(masks, rep);

    run_vit(3,  keep,    out_ctx);   // context ViT on masked patches
    run_vit(21, nullptr, out_tgt);   // target ViT on full images

    // predictor: relu(x@w1^T+b1)@w2^T+b2 on (ctx_emb + mask_token)
    addvec_k<<<(BATCH * DM + 255) / 256, 256, 0, stream>>>(out_ctx, mask_tok, ptmp0);
    cvt(pred_w1, (size_t)DM * DM);
    gemm(ptmp0, pred_b1, nullptr, ptmp1, BATCH, DM, DM, 2);
    cvt(pred_w2, (size_t)DM * DM);
    gemm(ptmp1, pred_b2, nullptr, ptmp0, BATCH, DM, DM, 0);
    // decoder -> (B, 192) patch
    cvt(dec_w1, (size_t)DM * DM);
    gemm(ptmp0, dec_b1, nullptr, ptmp1, BATCH, DM, DM, 2);
    cvt(dec_w2, (size_t)PATCHE * DM);
    gemm(ptmp1, dec_b2, nullptr, ppatch, BATCH, PATCHE, DM, 0);

    const int tot_img = BATCH * 3 * IMGS * IMGS;
    reconstruct_k<<<(tot_img + 255) / 256, 256, 0, stream>>>(images, ppatch, rep, out_rec);
}
